// PoseGNN_28776280883651
// MI455X (gfx1250) — matrix operand order repaired and checked by reading, not compiled, not run on hardware
//
#include <hip/hip_runtime.h>

// ---------------------------------------------------------------------------
// PoseGNN on MI455X (gfx1250): bf16 WMMA message-passing GNN.
//  - GEMMs computed flipped (D = Wt x actT): weights are the A operand,
//    row-major activation tiles are the B operand, so each lane's 8 D values
//    are 8 consecutive output features of one edge -> single b128 stores.
//  - Gathers use global_load_async_to_lds_b128 (ASYNCcnt path).
//  - Weight blob staged to LDS via TDM tensor_load_to_lds (6-arg builtin).
//  - Messages scatter-added with global_atomic_add_f32 into [N,128] f32
//    buffer (51.2 MB -> resident in the 192 MB L2).
// Workspace requirement: ~173 MB.
// ---------------------------------------------------------------------------

typedef __bf16 bf16;
typedef __attribute__((ext_vector_type(16))) __bf16 v16bf;
typedef __attribute__((ext_vector_type(8)))  __bf16 v8bf;
typedef __attribute__((ext_vector_type(8)))  float  v8f;
typedef __attribute__((ext_vector_type(4)))  unsigned int v4u;
typedef __attribute__((ext_vector_type(8)))  int v8i;
typedef __attribute__((ext_vector_type(4)))  int v4i;

#define E_CONST 1600000
#define N_CONST 100000

// Weight blob (in workspace) laid out so each kernel's LDS image is ONE
// contiguous slice (TDM-friendly):
//   [0      .. 114687] edge-step weights  (57344 bf16: EU0..EP1)
//   [114688 .. 116735] edge-step biases   (512 f32)
//   [116736 .. 159743] combine weights    (21504 bf16)
//   [159744 .. 160575] combine biases     (208 f32)
#define WU0 0
#define WU1 12288
#define WU2 18432
#define WF0 20480
#define WF1 32768
#define WP0 38912
#define WP1 51200
#define BU0 0
#define BU1 96
#define BU2 160
#define BF0 192
#define BF1 288
#define BP0 352
#define BP1 448
#define EDGE_SLICE_BYTES 116736
#define COMB_SLICE_BYTES 43840

#if defined(__has_builtin)
#if __has_builtin(__builtin_amdgcn_tensor_load_to_lds)
#define HAVE_TDM 1
#endif
#if __has_builtin(__builtin_amdgcn_s_wait_tensorcnt)
#define HAVE_TCNT 1
#endif
#endif

__device__ __forceinline__ void lds_fence() {
  asm volatile("s_wait_dscnt 0" ::: "memory");
}
__device__ __forceinline__ void wait_async() {
  asm volatile("s_wait_asynccnt 0" ::: "memory");
}

// Per-lane async global->LDS copy, n16 x 16 bytes. ldsOff is a byte offset
// within the workgroup's dynamic LDS allocation (LDS_BASE-relative).
__device__ __forceinline__ void async_gather(unsigned ldsOff,
                                             unsigned long long ga, int n16) {
#pragma unroll
  for (int i = 0; i < 6; ++i)
    if (i < n16)
      asm volatile("global_load_async_to_lds_b128 %0, %1, off"
                   :: "v"(ldsOff + (unsigned)(i * 16)),
                      "v"(ga + (unsigned long long)(i * 16))
                   : "memory");
}

#if HAVE_TDM
// 1-D TDM descriptor: copy `bytes` (multiple of 8) from global to LDS off 0.
__device__ __forceinline__ void tdm_load_1d(unsigned ldsOff, const void* gsrc,
                                            unsigned bytes) {
  unsigned long long ga = (unsigned long long)gsrc;
  unsigned n8 = bytes >> 3;  // 8-byte elements
  v4u g0 = {1u,                                   // count=1 (valid descriptor)
            ldsOff,                               // lds_addr
            (unsigned)ga,                         // global_addr[31:0]
            (unsigned)((ga >> 32) & 0x1FFFFFFu) | (2u << 30)};  // addr hi|type=2
  v8i g1 = {(int)(3u << 16),                      // data_size=3 (8B), mask=0
            (int)((n8 & 0xFFFFu) << 16),          // tensor_dim0[15:0]
            (int)(((n8 >> 16) & 0xFFFFu) | (1u << 16)),  // dim0 hi | tensor_dim1=1
            (int)(n8 << 16),                      // tile_dim0
            0,
            (int)n8,                              // tensor_dim0_stride lo
            0, 0};
  v4i z4 = {0, 0, 0, 0};
  v8i z8 = {0, 0, 0, 0, 0, 0, 0, 0};
  __builtin_amdgcn_tensor_load_to_lds(g0, g1, z4, z4, z8, 0);
}
#endif

__device__ __forceinline__ void wait_tensor() {
#if HAVE_TCNT
  __builtin_amdgcn_s_wait_tensorcnt(0);
#else
  asm volatile("s_wait_tensorcnt 0" ::: "memory");
#endif
}

// One dense layer on a 16-row tile, flipped: D[outf x edge] = Wt x actT.
//  actIn : LDS bf16 [16][din] row-major (B operand: lane = edge column)
//  wt    : LDS bf16 [dout][din]          (A operand: lane = out-feature row)
// Lane (r = lane&15, h = lane>>4) produces features f0..f0+7 (f0 = g+8h) of
// edge r -> packed single b128 stores.
__device__ __forceinline__ void wmma_layer(
    const bf16* actIn, const bf16* wt, const float* bias,
    int din, int dout, bool relu, int lane,
    bf16* outLds, int outStride,
    float* scatterBase, const int* rowNode, int scatCol,
    bf16* gOut, long gRowBase, int gStride)
{
  const int r = lane & 15;
  const int h = lane >> 4;
  int node = 0;
  if (scatterBase) node = rowNode[r];
  for (int g = 0; g < dout; g += 16) {
    v8f acc = {0.f, 0.f, 0.f, 0.f, 0.f, 0.f, 0.f, 0.f};
    for (int c = 0; c < din; c += 32) {
      v16bf a, b;
      // A (weights): row g+r, K = c + h*16 .. +15 (16 contiguous bf16)
      const bf16* ap = wt + (g + r) * din + c + h * 16;
#pragma unroll
      for (int i = 0; i < 16; ++i) a[i] = ap[i];
      // B (activations): col r, K = c + h*16 .. +15 (16 contiguous bf16)
      const bf16* bp = actIn + r * din + c + h * 16;
#pragma unroll
      for (int i = 0; i < 16; ++i) b[i] = bp[i];
      acc = __builtin_amdgcn_wmma_f32_16x16x32_bf16(
          false, a, false, b, (short)0, acc, false, false);
    }
    const int f0 = g + 8 * h;
    const float4* bp4 = (const float4*)(bias + f0);
    float4 b0 = bp4[0], b1 = bp4[1];
    const float bv[8] = {b0.x, b0.y, b0.z, b0.w, b1.x, b1.y, b1.z, b1.w};
    v8bf o;
#pragma unroll
    for (int v = 0; v < 8; ++v) {
      float f = acc[v] + bv[v];
      if (relu) f = fmaxf(f, 0.f);
      o[v] = (bf16)f;
      if (scatterBase)
        atomicAdd(scatterBase + (size_t)node * 128 + scatCol + f0 + v, f);
    }
    if (outLds)
      *(v4u*)(outLds + r * outStride + f0) = __builtin_bit_cast(v4u, o);
    if (gOut)
      *(v4u*)(gOut + (gRowBase + r) * (long)gStride + f0) =
          __builtin_bit_cast(v4u, o);
  }
}

// --------------------------- weight prep -----------------------------------
__global__ __launch_bounds__(256) void k_wprep(const float* __restrict__ W,
                                               const float* __restrict__ B,
                                               bf16* __restrict__ wt,
                                               float* __restrict__ bias,
                                               int din, int dout) {
  int i = blockIdx.x * 256 + threadIdx.x;
  if (i < din * dout) {
    int n = i / din, k = i % din;
    wt[i] = (bf16)W[(size_t)k * dout + n];   // transpose [din][dout]->[dout][din]
  }
  if (i < dout) bias[i] = B[i];
}

// --------------------------- scalar MLPs -----------------------------------
__global__ __launch_bounds__(256) void k_edge_enc(
    const float* __restrict__ attr,
    const float* __restrict__ W0, const float* __restrict__ B0,
    const float* __restrict__ W1, const float* __restrict__ B1,
    const float* __restrict__ W2, const float* __restrict__ B2,
    bf16* __restrict__ ea, int E) {
  int e = blockIdx.x * 256 + threadIdx.x;
  if (e >= E) return;
  float a0[4], a1[8], a2[16];
  for (int i = 0; i < 4; ++i) a0[i] = attr[(size_t)e * 4 + i];
  for (int o = 0; o < 8; ++o) {
    float s = B0[o];
    for (int k = 0; k < 4; ++k) s += a0[k] * W0[k * 8 + o];
    a1[o] = fmaxf(s, 0.f);
  }
  for (int o = 0; o < 16; ++o) {
    float s = B1[o];
    for (int k = 0; k < 8; ++k) s += a1[k] * W1[k * 16 + o];
    a2[o] = fmaxf(s, 0.f);
  }
  for (int o = 0; o < 32; ++o) {
    float s = B2[o];
    for (int k = 0; k < 16; ++k) s += a2[k] * W2[k * 32 + o];
    ea[(size_t)e * 32 + o] = (bf16)s;
  }
}

__global__ __launch_bounds__(256) void k_node_enc(
    const float* __restrict__ pose,
    const float* __restrict__ W0, const float* __restrict__ B0,
    const float* __restrict__ W1, const float* __restrict__ B1,
    const float* __restrict__ W2, const float* __restrict__ B2,
    bf16* __restrict__ ix, bf16* __restrict__ x,
    float* __restrict__ xenc, int N) {
  int n = blockIdx.x * 256 + threadIdx.x;
  if (n >= N) return;
  float a0[19], a1[24], a2[36];
  for (int i = 0; i < 19; ++i) a0[i] = pose[(size_t)n * 19 + i];
  for (int o = 0; o < 24; ++o) {
    float s = B0[o];
    for (int k = 0; k < 19; ++k) s += a0[k] * W0[k * 24 + o];
    a1[o] = fmaxf(s, 0.f);
  }
  for (int o = 0; o < 36; ++o) {
    float s = B1[o];
    for (int k = 0; k < 24; ++k) s += a1[k] * W1[k * 36 + o];
    a2[o] = fmaxf(s, 0.f);
  }
  for (int o = 0; o < 48; ++o) {
    float s = B2[o];
    for (int k = 0; k < 36; ++k) s += a2[k] * W2[k * 48 + o];
    ix[(size_t)n * 48 + o] = (bf16)s;
    x[(size_t)n * 48 + o]  = (bf16)s;
    xenc[(size_t)n * 48 + o] = s;      // x_enc output (f32)
  }
}

__global__ __launch_bounds__(256) void k_classifier(
    const bf16* __restrict__ ea,
    const float* __restrict__ W0, const float* __restrict__ B0,
    const float* __restrict__ W1, const float* __restrict__ B1,
    const float* __restrict__ W2, const float* __restrict__ B2,
    const float* __restrict__ W3, const float* __restrict__ B3,
    float* __restrict__ out, int E) {
  int e = blockIdx.x * 256 + threadIdx.x;
  if (e >= E) return;
  float a0[32], a1[16], a2[8], a3[4];
  for (int i = 0; i < 32; ++i) a0[i] = (float)ea[(size_t)e * 32 + i];
  for (int o = 0; o < 16; ++o) {
    float s = B0[o];
    for (int k = 0; k < 32; ++k) s += a0[k] * W0[k * 16 + o];
    a1[o] = fmaxf(s, 0.f);
  }
  for (int o = 0; o < 8; ++o) {
    float s = B1[o];
    for (int k = 0; k < 16; ++k) s += a1[k] * W1[k * 8 + o];
    a2[o] = fmaxf(s, 0.f);
  }
  for (int o = 0; o < 4; ++o) {
    float s = B2[o];
    for (int k = 0; k < 8; ++k) s += a2[k] * W2[k * 4 + o];
    a3[o] = fmaxf(s, 0.f);
  }
  float s = B3[0];
  for (int k = 0; k < 4; ++k) s += a3[k] * W3[k];
  out[e] = s;
}

// --------------------------- edge MP step (WMMA) ----------------------------
// 2 waves/block, one 16-edge tile per wave.
// LDS: 116,736B weights+biases (TDM) + 2 x 13,440B tile buffers = 143,616 B.
__global__ __launch_bounds__(64) void k_edge_mp(
    const int* __restrict__ src, const int* __restrict__ dst,
    const bf16* __restrict__ x, const bf16* __restrict__ ix,
    bf16* __restrict__ ea, float* __restrict__ msgs,
    const void* __restrict__ wsrc) {
  extern __shared__ char smem[];
  bf16*  wS = (bf16*)smem;                       // 57344 bf16
  float* bS = (float*)(smem + 114688);           // 512 f32
  const int tid = threadIdx.x;
  const int wave = tid >> 5, lane = tid & 31, r = lane & 15, h = lane >> 4;
#if HAVE_TDM
  if (wave == 0) { tdm_load_1d(0, wsrc, EDGE_SLICE_BYTES); wait_tensor(); }
#else
  {
    const uint4* gw = (const uint4*)wsrc;
    uint4* lw = (uint4*)smem;
    for (int i = tid; i < EDGE_SLICE_BYTES / 16; i += 64) lw[i] = gw[i];
  }
#endif
  __syncthreads();

  const unsigned wb = 116736u + (unsigned)wave * 13440u;  // per-wave LDS base
  bf16* in1 = (bf16*)(smem + wb);            // [16][128] (x_i | x_j | ea)
  bf16* in2 = (bf16*)(smem + wb + 4096);     // [16][128] msg-MLP input
  bf16* t96 = (bf16*)(smem + wb + 8192);     // [16][96]
  bf16* t64 = (bf16*)(smem + wb + 11264);    // [16][64]
  int*  sI  = (int*)(smem + wb + 13312);
  int*  dI  = (int*)(smem + wb + 13376);

  const long base = ((long)blockIdx.x * 2 + wave) * 16;
  if (lane < 16) { sI[lane] = src[base + lane]; dI[lane] = dst[base + lane]; }
  lds_fence();
  const long e = base + r;

  // gather [x_i | x_j | ea] straight into LDS (async, no VGPR round-trip)
  if (h == 0) {
    int d = dI[r];
    async_gather(wb + (unsigned)r * 256u,        (unsigned long long)(x + (long)d * 48), 6);
    async_gather(wb + (unsigned)r * 256u + 192u, (unsigned long long)(ea + e * 32), 4);
  } else {
    int s = sI[r];
    async_gather(wb + (unsigned)r * 256u + 96u,  (unsigned long long)(x + (long)s * 48), 6);
  }
  wait_async();

  // edge_update: 128 -> 96 -> 64 -> 32; upd goes to in2 cols 48..79 + global ea
  wmma_layer(in1, wS + WU0, bS + BU0, 128, 96, true,  lane, t96, 96, 0, 0, 0, 0, 0, 0);
  lds_fence();
  wmma_layer(t96, wS + WU1, bS + BU1,  96, 64, true,  lane, t64, 64, 0, 0, 0, 0, 0, 0);
  lds_fence();
  wmma_layer(t64, wS + WU2, bS + BU2,  64, 32, false, lane, in2 + 48, 128, 0, 0, 0,
             ea, base, 32);

  // create_future input: [x_i | upd | initial_x[dst]]
  if (h == 0) {
    const uint4* s4 = (const uint4*)(in1 + r * 128);
    uint4* d4 = (uint4*)(in2 + r * 128);
#pragma unroll
    for (int i = 0; i < 6; ++i) d4[i] = s4[i];
  } else {
    int d = dI[r];
    async_gather(wb + 4096u + (unsigned)r * 256u + 160u,
                 (unsigned long long)(ix + (long)d * 48), 6);
  }
  lds_fence();
  wait_async();
  wmma_layer(in2, wS + WF0, bS + BF0, 128, 96, true,  lane, t96, 96, 0, 0, 0, 0, 0, 0);
  lds_fence();
  wmma_layer(t96, wS + WF1, bS + BF1,  96, 64, false, lane, 0, 0, msgs, sI, 64, 0, 0, 0);

  // create_past input: [x_j | upd | initial_x[src]]
  if (h == 0) {
    const uint4* s4 = (const uint4*)(in1 + r * 128 + 48);
    uint4* d4 = (uint4*)(in2 + r * 128);
#pragma unroll
    for (int i = 0; i < 6; ++i) d4[i] = s4[i];
  } else {
    int s = sI[r];
    async_gather(wb + 4096u + (unsigned)r * 256u + 160u,
                 (unsigned long long)(ix + (long)s * 48), 6);
  }
  lds_fence();
  wait_async();
  wmma_layer(in2, wS + WP0, bS + BP0, 128, 96, true,  lane, t96, 96, 0, 0, 0, 0, 0, 0);
  lds_fence();
  wmma_layer(t96, wS + WP1, bS + BP1,  96, 64, false, lane, 0, 0, msgs, dI, 0, 0, 0, 0);
}

// --------------------------- combine (WMMA) ---------------------------------
// 4 waves/block, one 16-node tile per wave.
// LDS: 43,840B weights+biases + 4 x 9,216B buffers = 80,704 B.
__global__ __launch_bounds__(128) void k_combine(
    const float* __restrict__ msgs, const void* __restrict__ wsrc,
    bf16* __restrict__ x, int N) {
  extern __shared__ char smem[];
  bf16*  wS = (bf16*)smem;                       // 21504 bf16
  float* bS = (float*)(smem + 43008);            // 208 f32
  const int tid = threadIdx.x;
  const int wave = tid >> 5, lane = tid & 31, r = lane & 15, h = lane >> 4;
#if HAVE_TDM
  if (wave == 0) { tdm_load_1d(0, wsrc, COMB_SLICE_BYTES); wait_tensor(); }
#else
  {
    const uint4* gw = (const uint4*)wsrc;
    uint4* lw = (uint4*)smem;
    for (int i = tid; i < COMB_SLICE_BYTES / 16; i += 128) lw[i] = gw[i];
  }
#endif
  __syncthreads();

  const long tiles = N / 16;
  const long tile = (long)blockIdx.x * 4 + wave;
  if (tile >= tiles) return;
  const unsigned wb = 43840u + (unsigned)wave * 9216u;
  bf16* inW = (bf16*)(smem + wb);            // [16][128]
  bf16* t96 = (bf16*)(smem + wb + 4096);
  bf16* t64 = (bf16*)(smem + wb + 7168);

  const long n0 = tile * 16;
  {
    const long n = n0 + r;
    const float4* mrow = (const float4*)(msgs + (size_t)n * 128 + h * 64);
    bf16* irow = inW + r * 128 + h * 64;
#pragma unroll
    for (int i = 0; i < 8; ++i) {
      float4 va = mrow[2 * i], vb = mrow[2 * i + 1];
      v8bf o;
      o[0] = (bf16)va.x; o[1] = (bf16)va.y; o[2] = (bf16)va.z; o[3] = (bf16)va.w;
      o[4] = (bf16)vb.x; o[5] = (bf16)vb.y; o[6] = (bf16)vb.z; o[7] = (bf16)vb.w;
      *(v4u*)(irow + i * 8) = __builtin_bit_cast(v4u, o);
    }
  }
  lds_fence();
  // combine: 128 -> 96 -> 64 -> 48 (no relu on last; written straight to x)
  wmma_layer(inW, wS + 0,     bS + 0,   128, 96, true,  lane, t96, 96, 0, 0, 0, 0, 0, 0);
  lds_fence();
  wmma_layer(t96, wS + 12288, bS + 96,   96, 64, true,  lane, t64, 64, 0, 0, 0, 0, 0, 0);
  lds_fence();
  wmma_layer(t64, wS + 18432, bS + 160,  64, 48, false, lane, 0, 0, 0, 0, 0, x, n0, 48);
}

// --------------------------- host driver ------------------------------------
extern "C" void kernel_launch(void* const* d_in, const int* in_sizes, int n_in,
                              void* d_out, int out_size, void* d_ws, size_t ws_size,
                              hipStream_t stream) {
  if (n_in < 44) return;
  const int E = E_CONST, N = N_CONST;
  const float* pose  = (const float*)d_in[0];
  const int*   ei    = (const int*)d_in[1];
  const int*   srcI  = ei;
  const int*   dstI  = ei + E;
  const float* eattr = (const float*)d_in[2];

  // Param leaf order: insertion (setup_inputs dict) vs JAX pytree (sorted).
  const bool ins = (in_sizes[4] == 4 * 8);
  int eu[6], cpm[4], cfm[4], cbm[6], eem[6], nem[6], ecm[8];
  if (ins) {
    for (int i = 0; i < 6; ++i) eem[i] = 4 + i;
    for (int i = 0; i < 6; ++i) nem[i] = 10 + i;
    for (int i = 0; i < 8; ++i) ecm[i] = 16 + i;
    for (int i = 0; i < 6; ++i) eu[i]  = 24 + i;
    for (int i = 0; i < 4; ++i) cpm[i] = 30 + i;
    for (int i = 0; i < 4; ++i) cfm[i] = 34 + i;
    for (int i = 0; i < 6; ++i) cbm[i] = 38 + i;
  } else {
    for (int i = 0; i < 6; ++i) cbm[i] = 4 + i;
    for (int i = 0; i < 4; ++i) cfm[i] = 10 + i;
    for (int i = 0; i < 4; ++i) cpm[i] = 14 + i;
    for (int i = 0; i < 8; ++i) ecm[i] = 18 + i;
    for (int i = 0; i < 6; ++i) eem[i] = 26 + i;
    for (int i = 0; i < 6; ++i) eu[i]  = 32 + i;
    for (int i = 0; i < 6; ++i) nem[i] = 38 + i;
  }
  auto P = [&](int i) { return (const float*)d_in[i]; };

  // workspace layout (bytes)
  char* ws = (char*)d_ws;
  bf16*  ea    = (bf16*)(ws + 0);                   // E*32 bf16 = 102.4 MB
  bf16*  ix    = (bf16*)(ws + 102400000);           // N*48 bf16
  bf16*  x     = (bf16*)(ws + 112000000);           // N*48 bf16
  float* msgs  = (float*)(ws + 121600000);          // N*128 f32 = 51.2 MB
  char*  wsW   = ws + 172800000;                    // weight blob (160,576 B)
  bf16*  wE = (bf16*)wsW;                           // edge weights
  float* bE = (float*)(wsW + 114688);               // edge biases
  bf16*  wC = (bf16*)(wsW + 116736);                // combine weights
  float* bC = (float*)(wsW + 159744);               // combine biases

  // convert + transpose the 10 WMMA-consumed layers to bf16
  bf16*  wdst[10] = {wE + WU0, wE + WU1, wE + WU2, wE + WF0, wE + WF1,
                     wE + WP0, wE + WP1, wC + 0, wC + 12288, wC + 18432};
  float* bdst[10] = {bE + BU0, bE + BU1, bE + BU2, bE + BF0, bE + BF1,
                     bE + BP0, bE + BP1, bC + 0, bC + 96, bC + 160};
  const int wi[10] = {eu[0], eu[2], eu[4], cfm[0], cfm[2], cpm[0], cpm[2],
                      cbm[0], cbm[2], cbm[4]};
  const int bi[10] = {eu[1], eu[3], eu[5], cfm[1], cfm[3], cpm[1], cpm[3],
                      cbm[1], cbm[3], cbm[5]};
  const int dins[10]  = {128, 96, 64, 128, 96, 128, 96, 128, 96, 64};
  const int douts[10] = {96, 64, 32, 96, 64, 96, 64, 96, 64, 48};
  for (int l = 0; l < 10; ++l) {
    int total = dins[l] * douts[l];
    k_wprep<<<(total + 255) / 256, 256, 0, stream>>>(
        P(wi[l]), P(bi[l]), wdst[l], bdst[l], dins[l], douts[l]);
  }

  // encoders
  k_edge_enc<<<(E + 255) / 256, 256, 0, stream>>>(
      eattr, P(eem[0]), P(eem[1]), P(eem[2]), P(eem[3]), P(eem[4]), P(eem[5]),
      ea, E);
  k_node_enc<<<(N + 255) / 256, 256, 0, stream>>>(
      pose, P(nem[0]), P(nem[1]), P(nem[2]), P(nem[3]), P(nem[4]), P(nem[5]),
      ix, x, (float*)d_out + E, N);

  // 6 message-passing steps
  const int gridE = E / 32;                 // 2 tiles of 16 edges per block
  const int gridC = (N / 16 + 3) / 4;       // 4 tiles of 16 nodes per block
  for (int step = 0; step < 6; ++step) {
    (void)hipMemsetAsync(msgs, 0, (size_t)N * 128 * sizeof(float), stream);
    k_edge_mp<<<gridE, 64, 143616, stream>>>(srcI, dstI, x, ix, ea, msgs, wsW);
    k_combine<<<gridC, 128, 80704, stream>>>(msgs, wsW + 116736, x, N);
  }

  // classifier on final edge features
  k_classifier<<<(E + 255) / 256, 256, 0, stream>>>(
      ea, P(ecm[0]), P(ecm[1]), P(ecm[2]), P(ecm[3]), P(ecm[4]), P(ecm[5]),
      P(ecm[6]), P(ecm[7]), (float*)d_out, E);
}
